// MyLSTMLayer_34995393528349
// MI455X (gfx1250) — compile-verified, matrix operand order checked by reading, not run on hardware
//
#include <hip/hip_runtime.h>

// ---------------------------------------------------------------------------
// LSTM layer, MI455X (gfx1250, wave32).
//   Phase 1 (parallel):  Xpre[g] = X @ Wg[:, :IN]^T  via v_wmma_f32_16x16x32_bf16
//     - 128x64 output tile per 256-thread block, K-chunk 64, LDS double buffer
//       (B tile shared by 8 waves -> ~3.5x less L2 traffic than per-wave loads)
//   Phase 2 (sequential): persistent kernel, per step t:
//       gates  = sigmoid/tanh( Xpre[g][t] + Wg[:,IN:] @ h + bg )   (bf16 wts, f32 acc)
//       c      = f*c + i*cn
//       h      = relu( Wh @ c + bh );  out[t] = h
//     grid-wide sense barrier between the two sub-steps.
// Recurrent weights (32 MB bf16) are L2-resident (192 MB L2) after step 0.
// ---------------------------------------------------------------------------

#define SEQ 4096
#define INF_ 2048
#define HID 2048
#define COMB (INF_ + HID)

typedef __bf16        v16bf __attribute__((ext_vector_type(16)));
typedef float         v8f   __attribute__((ext_vector_type(8)));
typedef unsigned int  u32x4 __attribute__((ext_vector_type(4)));

union Frag { u32x4 u[2]; v16bf v; };

// ---- bf16 helpers (storage = unsigned short) ------------------------------
__device__ __forceinline__ unsigned short f2bf(float f) {
  unsigned u = __float_as_uint(f);
  unsigned r = u + 0x7FFFu + ((u >> 16) & 1u);   // round-to-nearest-even
  return (unsigned short)(r >> 16);
}
__device__ __forceinline__ float bflo(unsigned w) { return __uint_as_float(w << 16); }
__device__ __forceinline__ float bfhi(unsigned w) { return __uint_as_float(w & 0xFFFF0000u); }

__device__ __forceinline__ float wave_sum(float v) {
  #pragma unroll
  for (int off = 16; off > 0; off >>= 1) v += __shfl_xor(v, off, 32);
  return v;
}

// ---------------------------------------------------------------------------
// init: zero h, c, barrier state (ws is poisoned 0xAA; must re-init each call)
// ---------------------------------------------------------------------------
__global__ void k_init(float* __restrict__ h, float* __restrict__ c,
                       unsigned* __restrict__ bar) {
  int i = blockIdx.x * blockDim.x + threadIdx.x;
  if (i < HID) { h[i] = 0.f; c[i] = 0.f; }
  if (i == 0) { bar[0] = 0u; bar[1] = 0u; }
}

// ---------------------------------------------------------------------------
// fp32 -> bf16 conversions
// ---------------------------------------------------------------------------
__global__ void k_cvt(const float* __restrict__ src, unsigned short* __restrict__ dst,
                      int n) {
  int i = blockIdx.x * blockDim.x + threadIdx.x;
  if (i < n) dst[i] = f2bf(src[i]);
}

// split one gate weight [HID][IN+HID] -> Wx [HID][IN] + Wh_part [HID][HID]
__global__ void k_cvt_gate(const float* __restrict__ W,
                           unsigned short* __restrict__ Wx,
                           unsigned short* __restrict__ Whp) {
  size_t i = (size_t)blockIdx.x * blockDim.x + threadIdx.x;
  if (i >= (size_t)HID * COMB) return;
  int j = (int)(i / COMB);
  int k = (int)(i % COMB);
  unsigned short v = f2bf(W[i]);
  if (k < INF_) Wx[(size_t)j * INF_ + k] = v;
  else          Whp[(size_t)j * HID + (k - INF_)] = v;
}

// ---------------------------------------------------------------------------
// Phase 1 GEMM: Xpre[g][t][j] = sum_k Xb[t][k] * Wxb[g][j][k]   (bf16 -> f32)
// Block = 256 threads (8 waves) -> 128(t) x 64(j) output tile.
// Wave w computes t-subtile w*16 and 4 j-tiles; A/B staged in LDS, double-buf.
// ---------------------------------------------------------------------------
#define TBL 128
#define JBL 64
#define KC  64
#define CH  (INF_ / KC)      // 32 K-chunks
#define GEMM_BLOCKS (3 * (SEQ / TBL) * (HID / JBL))   // 3 * 32 * 32 = 3072

__global__ void __launch_bounds__(256)
k_gemm(const unsigned short* __restrict__ Xb,
       const unsigned short* __restrict__ Wxb,
       float* __restrict__ Xpre) {
  constexpr int LDA = 72;   // padded row stride (elements) to spread LDS banks
  constexpr int LDB = 72;
  __shared__ __align__(16) unsigned short shA[2][TBL * LDA];  // 2 x 18 KB
  __shared__ __align__(16) unsigned short shB[2][JBL * LDB];  // 2 x  9 KB

  const int tid  = threadIdx.x;
  const int lane = tid & 31;
  const int wave = tid >> 5;

  const int per_gate = (SEQ / TBL) * (HID / JBL);
  const int gate = blockIdx.x / per_gate;
  int rem = blockIdx.x % per_gate;
  const int t0 = (rem / (HID / JBL)) * TBL;
  const int j0 = (rem % (HID / JBL)) * JBL;

  const unsigned short* wg = Wxb + (size_t)gate * HID * INF_;

  // staging map: each thread moves uint4 (8 bf16); row = tid>>3 (+p*32)
  const int srow = tid >> 3;           // 0..31
  const int sqc  = (tid & 7) * 8;      // element offset within K-chunk

  const unsigned short* aG[4];
  const unsigned short* bG[2];
  #pragma unroll
  for (int p = 0; p < 4; ++p)
    aG[p] = Xb + (size_t)(t0 + srow + p * 32) * INF_ + sqc;
  #pragma unroll
  for (int p = 0; p < 2; ++p)
    bG[p] = wg + (size_t)(j0 + srow + p * 32) * INF_ + sqc;

  // stage K-chunk 0 into buffer 0
  #pragma unroll
  for (int p = 0; p < 4; ++p)
    *(u32x4*)&shA[0][(srow + p * 32) * LDA + sqc] = *(const u32x4*)(aG[p]);
  #pragma unroll
  for (int p = 0; p < 2; ++p)
    *(u32x4*)&shB[0][(srow + p * 32) * LDB + sqc] = *(const u32x4*)(bG[p]);
  __syncthreads();

  const int mrow  = lane & 15;
  const int kbase = (lane >> 4) * 8;   // ISA 16-bit A-fragment K layout

  v8f acc0 = {}, acc1 = {}, acc2 = {}, acc3 = {};

  for (int ch = 0; ch < CH; ++ch) {
    const int cur = ch & 1;
    const int nxt = cur ^ 1;
    const bool more = (ch + 1) < CH;

    // issue next chunk's global loads early; they drain while we do LDS+WMMA
    u32x4 ra[4], rb[2];
    if (more) {
      const int koff = (ch + 1) * KC;
      #pragma unroll
      for (int p = 0; p < 4; ++p) ra[p] = *(const u32x4*)(aG[p] + koff);
      #pragma unroll
      for (int p = 0; p < 2; ++p) rb[p] = *(const u32x4*)(bG[p] + koff);
    }

    #pragma unroll
    for (int ks = 0; ks < KC; ks += 32) {
      Frag a;
      const unsigned short* abase =
          &shA[cur][(wave * 16 + mrow) * LDA + ks + kbase];
      a.u[0] = *(const u32x4*)(abase);        // K = kbase .. kbase+7
      a.u[1] = *(const u32x4*)(abase + 16);   // K = kbase+16 .. +23
      #pragma unroll
      for (int jt = 0; jt < 4; ++jt) {
        Frag b;
        const unsigned short* bbase =
            &shB[cur][(jt * 16 + mrow) * LDB + ks + kbase];
        b.u[0] = *(const u32x4*)(bbase);
        b.u[1] = *(const u32x4*)(bbase + 16);
        v8f& acc = (jt == 0) ? acc0 : (jt == 1) ? acc1 : (jt == 2) ? acc2 : acc3;
        acc = __builtin_amdgcn_wmma_f32_16x16x32_bf16(
            /*neg_a=*/false, a.v, /*neg_b=*/false, b.v,
            /*c_mod=*/(short)0, acc, /*reuse_a=*/false, /*reuse_b=*/false);
      }
    }

    if (more) {
      #pragma unroll
      for (int p = 0; p < 4; ++p)
        *(u32x4*)&shA[nxt][(srow + p * 32) * LDA + sqc] = ra[p];
      #pragma unroll
      for (int p = 0; p < 2; ++p)
        *(u32x4*)&shB[nxt][(srow + p * 32) * LDB + sqc] = rb[p];
    }
    __syncthreads();
  }

  // D layout: lanes 0-15 -> M = v, N = lane; lanes 16-31 -> M = 8+v, N = lane-16
  const int rbase = (lane >> 4) * 8;
  float* outg = Xpre + (size_t)gate * SEQ * HID;
  const int trow0 = t0 + wave * 16 + rbase;
  #pragma unroll
  for (int v = 0; v < 8; ++v) {
    size_t row = (size_t)(trow0 + v) * HID;
    outg[row + j0 +  0 + mrow] = acc0[v];
    outg[row + j0 + 16 + mrow] = acc1[v];
    outg[row + j0 + 32 + mrow] = acc2[v];
    outg[row + j0 + 48 + mrow] = acc3[v];
  }
}

// ---------------------------------------------------------------------------
// Phase 2: persistent sequential LSTM kernel with grid-wide barrier
// ---------------------------------------------------------------------------
#define NBLK 64
#define NTHR 256

__device__ __forceinline__ void grid_barrier(unsigned* cnt, unsigned* gen,
                                             unsigned nblocks) {
  __syncthreads();
  if (threadIdx.x == 0) {
    __threadfence();
    unsigned g = __hip_atomic_load(gen, __ATOMIC_RELAXED, __HIP_MEMORY_SCOPE_AGENT);
    unsigned arrived = __hip_atomic_fetch_add(cnt, 1u, __ATOMIC_ACQ_REL,
                                              __HIP_MEMORY_SCOPE_AGENT) + 1u;
    if (arrived == nblocks) {
      __hip_atomic_store(cnt, 0u, __ATOMIC_RELAXED, __HIP_MEMORY_SCOPE_AGENT);
      __hip_atomic_fetch_add(gen, 1u, __ATOMIC_RELEASE, __HIP_MEMORY_SCOPE_AGENT);
    } else {
      while (__hip_atomic_load(gen, __ATOMIC_ACQUIRE, __HIP_MEMORY_SCOPE_AGENT) == g) {
        __builtin_amdgcn_s_sleep(2);
      }
    }
    __threadfence();
  }
  __syncthreads();
}

__global__ void __launch_bounds__(NTHR)
k_lstm(const float* __restrict__ Xpre,
       const unsigned short* __restrict__ Wfh,
       const unsigned short* __restrict__ Wih,
       const unsigned short* __restrict__ Wch,
       const unsigned short* __restrict__ Whb,
       const float* __restrict__ bf, const float* __restrict__ bi,
       const float* __restrict__ bc, const float* __restrict__ bh,
       float* __restrict__ h, float* __restrict__ c,
       float* __restrict__ out, unsigned* __restrict__ bar) {
  __shared__ float sh[HID];
  const int tid  = threadIdx.x;
  const int lane = tid & 31;
  const int wid  = blockIdx.x * (NTHR >> 5) + (tid >> 5);
  const int nw   = NBLK * (NTHR >> 5);            // 512 waves
  unsigned* cnt = bar;
  unsigned* gen = bar + 1;

  for (int t = 0; t < SEQ; ++t) {
    // ---- stage 1: gates + cell update --------------------------------
    for (int k = tid; k < HID; k += NTHR) sh[k] = h[k];
    __syncthreads();
    const float* xpf = Xpre + ((size_t)0 * SEQ + t) * HID;
    const float* xpi = Xpre + ((size_t)1 * SEQ + t) * HID;
    const float* xpc = Xpre + ((size_t)2 * SEQ + t) * HID;

    for (int j = wid; j < HID; j += nw) {
      const unsigned* wf = (const unsigned*)(Wfh + (size_t)j * HID);
      const unsigned* wi = (const unsigned*)(Wih + (size_t)j * HID);
      const unsigned* wc = (const unsigned*)(Wch + (size_t)j * HID);
      float af = 0.f, ai = 0.f, ac = 0.f;
      for (int k = lane * 2; k < HID; k += 64) {
        float2 hv = *(const float2*)&sh[k];
        int ki = k >> 1;
        unsigned uf = wf[ki]; af += hv.x * bflo(uf) + hv.y * bfhi(uf);
        unsigned ui = wi[ki]; ai += hv.x * bflo(ui) + hv.y * bfhi(ui);
        unsigned uc = wc[ki]; ac += hv.x * bflo(uc) + hv.y * bfhi(uc);
      }
      af = wave_sum(af); ai = wave_sum(ai); ac = wave_sum(ac);
      if (lane == 0) {
        float fpre = af + bf[j] + xpf[j];
        float ipre = ai + bi[j] + xpi[j];
        float cpre = ac + bc[j] + xpc[j];
        float fg = 1.f / (1.f + __expf(-fpre));
        float ig = 1.f / (1.f + __expf(-ipre));
        float cn = tanhf(cpre);
        c[j] = fg * c[j] + ig * cn;
      }
    }
    grid_barrier(cnt, gen, NBLK);

    // ---- stage 2: h = relu(Wh @ c + bh) ------------------------------
    for (int k = tid; k < HID; k += NTHR) sh[k] = c[k];
    __syncthreads();
    for (int j = wid; j < HID; j += nw) {
      const unsigned* wr = (const unsigned*)(Whb + (size_t)j * HID);
      float a = 0.f;
      for (int k = lane * 2; k < HID; k += 64) {
        float2 cv = *(const float2*)&sh[k];
        unsigned u = wr[k >> 1];
        a += cv.x * bflo(u) + cv.y * bfhi(u);
      }
      a = wave_sum(a);
      if (lane == 0) {
        float hv = a + bh[j];
        hv = hv > 0.f ? hv : 0.f;
        h[j] = hv;
        out[(size_t)t * HID + j] = hv;
      }
    }
    grid_barrier(cnt, gen, NBLK);
  }
}

// ---------------------------------------------------------------------------
// workspace layout (all offsets 256B aligned)
// ---------------------------------------------------------------------------
static constexpr size_t OFF_XB   = 0;                                   // 16 MB bf16 X
static constexpr size_t OFF_WXB  = OFF_XB  + (size_t)SEQ * INF_ * 2;    // 24 MB bf16 Wx[3]
static constexpr size_t OFF_WHP  = OFF_WXB + (size_t)3 * HID * INF_ * 2;// 24 MB bf16 Wgh[3]
static constexpr size_t OFF_WHB  = OFF_WHP + (size_t)3 * HID * HID * 2; //  8 MB bf16 Wh
static constexpr size_t OFF_XPRE = OFF_WHB + (size_t)HID * HID * 2;     // 96 MB f32 Xpre[3]
static constexpr size_t OFF_H    = OFF_XPRE + (size_t)3 * SEQ * HID * 4;
static constexpr size_t OFF_C    = OFF_H + (size_t)HID * 4;
static constexpr size_t OFF_BAR  = OFF_C + (size_t)HID * 4;

extern "C" void kernel_launch(void* const* d_in, const int* in_sizes, int n_in,
                              void* d_out, int out_size, void* d_ws, size_t ws_size,
                              hipStream_t stream) {
  (void)in_sizes; (void)n_in; (void)out_size; (void)ws_size;
  const float* x  = (const float*)d_in[0];
  const float* Wf = (const float*)d_in[1];
  const float* bf = (const float*)d_in[2];
  const float* Wi = (const float*)d_in[3];
  const float* bi = (const float*)d_in[4];
  const float* Wc = (const float*)d_in[5];
  const float* bc = (const float*)d_in[6];
  const float* Wh = (const float*)d_in[7];
  const float* bh = (const float*)d_in[8];
  float* out = (float*)d_out;

  char* ws = (char*)d_ws;
  unsigned short* Xb   = (unsigned short*)(ws + OFF_XB);
  unsigned short* Wxb  = (unsigned short*)(ws + OFF_WXB);
  unsigned short* Whp  = (unsigned short*)(ws + OFF_WHP);
  unsigned short* Whb  = (unsigned short*)(ws + OFF_WHB);
  float*          Xpre = (float*)(ws + OFF_XPRE);
  float*          hbuf = (float*)(ws + OFF_H);
  float*          cbuf = (float*)(ws + OFF_C);
  unsigned*       bar  = (unsigned*)(ws + OFF_BAR);

  k_init<<<8, 256, 0, stream>>>(hbuf, cbuf, bar);

  // conversions
  {
    int n = SEQ * INF_;
    k_cvt<<<(n + 255) / 256, 256, 0, stream>>>(x, Xb, n);
  }
  {
    int n = HID * HID;
    k_cvt<<<(n + 255) / 256, 256, 0, stream>>>(Wh, Whb, n);
  }
  {
    size_t n = (size_t)HID * COMB;
    int blocks = (int)((n + 255) / 256);
    k_cvt_gate<<<blocks, 256, 0, stream>>>(Wf, Wxb + (size_t)0 * HID * INF_,
                                           Whp + (size_t)0 * HID * HID);
    k_cvt_gate<<<blocks, 256, 0, stream>>>(Wi, Wxb + (size_t)1 * HID * INF_,
                                           Whp + (size_t)1 * HID * HID);
    k_cvt_gate<<<blocks, 256, 0, stream>>>(Wc, Wxb + (size_t)2 * HID * INF_,
                                           Whp + (size_t)2 * HID * HID);
  }

  // phase 1: WMMA GEMM for the x-dependent gate pre-activations
  k_gemm<<<GEMM_BLOCKS, 256, 0, stream>>>(Xb, Wxb, Xpre);

  // phase 2: persistent sequential recurrence
  k_lstm<<<NBLK, NTHR, 0, stream>>>(Xpre,
                                    Whp + (size_t)0 * HID * HID,
                                    Whp + (size_t)1 * HID * HID,
                                    Whp + (size_t)2 * HID * HID,
                                    Whb, bf, bi, bc, bh,
                                    hbuf, cbuf, out, bar);
}
